// CEBlock_88983132439287
// MI455X (gfx1250) — compile-verified, matrix operand order checked by reading, not run on hardware
//
#include <hip/hip_runtime.h>
#include <hip/hip_bf16.h>
#include <cstdint>

// ---------------- CDNA5 WMMA primitives ----------------
typedef __attribute__((ext_vector_type(16))) _Float16 v16h;
typedef __attribute__((ext_vector_type(8)))  float    v8f;

union Frag16 { v16h h; float4 f[2]; };

// Load a 16x32 (rows x K) f16 fragment per ISA 7.12.2:
//  lane&15 = row; lanes 0-15 hold K 0..7 & 16..23; lanes 16-31 hold K 8..15 & 24..31.
__device__ __forceinline__ v16h load_ab_frag(const _Float16* tile, int ld) {
    const int lane = threadIdx.x & 31;
    const _Float16* p = tile + (size_t)(lane & 15) * ld + ((lane >> 4) << 3);
    Frag16 fr;
    fr.f[0] = *(const float4*)(p);
    fr.f[1] = *(const float4*)(p + 16);
    return fr.h;
}

__device__ __forceinline__ v8f wmma_f16(v16h a, v16h b, v8f c) {
    return __builtin_amdgcn_wmma_f32_16x16x32_f16(false, a, false, b, (short)0, c, false, false);
}

// ---------------- problem constants ----------------
#define DIMC   768
#define HEADS  12
#define HD     64
#define LT     64
#define LS     256
#define NTOK   576          // LT + 2*LS
#define BATCH  32
#define MROWS  (BATCH*NTOK) // 18432
#define MLPH   3072
#define KEEPN  180          // ceil(0.7*256)
#define REMN   76

// ---------------- generic WMMA GEMM: C[M,N] = A[M,K] * W[N,K]^T (+epilogue) --------
// Per-wave 64x32 tile (4x2 WMMA tiles): 8 WMMAs per 6 fragment loads
// => 21.3 FLOP/byte register-level intensity with only ~130 VGPRs (no spills).
struct EpiArgs {
    const float* bias;
    const float* resid;
    float*       out_f32;
    _Float16*    oh0;   // q (or f16 out)
    _Float16*    oh1;   // k
    _Float16*    oh2;   // v^T
};

enum { EPI_QKV = 0, EPI_GELU16 = 1, EPI_RESID32 = 2 };

template <int EPI>
__global__ __launch_bounds__(256) void gemm_wmma_kernel(
    const _Float16* __restrict__ A, const _Float16* __restrict__ W,
    int M, int N, int K, EpiArgs ea)
{
    const int lane = threadIdx.x & 31;
    const int wave = threadIdx.x >> 5;
    const int wm = wave & 3;                 // 4 waves along M
    const int wn = wave >> 2;                // 2 waves along N
    const int m0 = blockIdx.x * 256 + wm * 64;
    const int n0 = blockIdx.y * 64  + wn * 32;

    v8f acc[4][2];
#pragma unroll
    for (int i = 0; i < 4; ++i)
#pragma unroll
        for (int j = 0; j < 2; ++j) acc[i][j] = (v8f){0,0,0,0,0,0,0,0};

    for (int k = 0; k < K; k += 32) {
        v16h a[4];
#pragma unroll
        for (int i = 0; i < 4; ++i)
            a[i] = load_ab_frag(A + (size_t)(m0 + 16 * i) * K + k, K);
#pragma unroll
        for (int j = 0; j < 2; ++j) {
            v16h b = load_ab_frag(W + (size_t)(n0 + 16 * j) * K + k, K);
#pragma unroll
            for (int i = 0; i < 4; ++i)
                acc[i][j] = wmma_f16(a[i], b, acc[i][j]);
        }
    }

    // ---- epilogue: all div/mod hoisted out of the per-element loop ----
    const int cn = lane & 15;
    const int rb = (lane >> 4) << 3;     // 0 or 8
#pragma unroll
    for (int j = 0; j < 2; ++j) {
        const int   col  = n0 + j * 16 + cn;
        const float bcol = ea.bias[col];
        if (EPI == EPI_QKV) {
            const int t  = col / DIMC;           // constant per j-tile
            const int hh = (col % DIMC) / HD;
            const int dd = col & (HD - 1);
#pragma unroll
            for (int i = 0; i < 4; ++i) {
                const int r0 = m0 + i * 16;      // 16-row tile within one batch
                const int b  = r0 / NTOK;
                const int nb = (r0 % NTOK) + rb;
                if (t == 2) {
                    // v^T: [b][h][dd][token], stride 1 along e
                    _Float16* p = ea.oh2 + (((size_t)(b*HEADS+hh))*HD + dd)*NTOK + nb;
#pragma unroll
                    for (int e = 0; e < 8; ++e)
                        p[e] = (_Float16)(acc[i][j][e] + bcol);
                } else {
                    // q/k: [b][h][token][dd], stride HD along e
                    _Float16* p = (t == 0 ? ea.oh0 : ea.oh1)
                                + (((size_t)(b*HEADS+hh))*NTOK + nb)*HD + dd;
#pragma unroll
                    for (int e = 0; e < 8; ++e)
                        p[(size_t)e * HD] = (_Float16)(acc[i][j][e] + bcol);
                }
            }
        } else if (EPI == EPI_GELU16) {
#pragma unroll
            for (int i = 0; i < 4; ++i) {
                _Float16* p = ea.oh0 + (size_t)(m0 + i * 16 + rb) * N + col;
#pragma unroll
                for (int e = 0; e < 8; ++e) {
                    const float v = acc[i][j][e] + bcol;
                    p[(size_t)e * N] =
                        (_Float16)(0.5f * v * (1.0f + erff(v * 0.7071067811865475f)));
                }
            }
        } else { // EPI_RESID32
#pragma unroll
            for (int i = 0; i < 4; ++i) {
                const size_t base = (size_t)(m0 + i * 16 + rb) * N + col;
#pragma unroll
                for (int e = 0; e < 8; ++e)
                    ea.out_f32[base + (size_t)e * N] =
                        ea.resid[base + (size_t)e * N] + acc[i][j][e] + bcol;
            }
        }
    }
}

// ---------------- LayerNorm (f32 in -> f16 out), one block per token row ----------
__global__ __launch_bounds__(256) void ln_kernel(
    const float* __restrict__ x, const float* __restrict__ w,
    const float* __restrict__ bb, _Float16* __restrict__ out)
{
    __shared__ float red[8];
    const int row = blockIdx.x;
    const float* xr = x + (size_t)row * DIMC;
    const int t = threadIdx.x, lane = t & 31, wid = t >> 5;

    float v0 = xr[t], v1 = xr[t + 256], v2 = xr[t + 512];
    float s = v0 + v1 + v2;
    for (int o = 16; o; o >>= 1) s += __shfl_xor(s, o, 32);
    if (lane == 0) red[wid] = s;
    __syncthreads();
    float mean = 0.f;
#pragma unroll
    for (int i = 0; i < 8; ++i) mean += red[i];
    mean *= (1.0f / DIMC);
    __syncthreads();

    float d0 = v0 - mean, d1 = v1 - mean, d2 = v2 - mean;
    float vs = d0*d0 + d1*d1 + d2*d2;
    for (int o = 16; o; o >>= 1) vs += __shfl_xor(vs, o, 32);
    if (lane == 0) red[wid] = vs;
    __syncthreads();
    float var = 0.f;
#pragma unroll
    for (int i = 0; i < 8; ++i) var += red[i];
    var *= (1.0f / DIMC);
    const float rstd = rsqrtf(var + 1e-5f);

    _Float16* orow = out + (size_t)row * DIMC;
    orow[t]       = (_Float16)(d0 * rstd * w[t]       + bb[t]);
    orow[t + 256] = (_Float16)(d1 * rstd * w[t + 256] + bb[t + 256]);
    orow[t + 512] = (_Float16)(d2 * rstd * w[t + 512] + bb[t + 512]);
}

// ---------------- fused attention: S = QK^T, softmax, attn out, score acc, O = PV --
__global__ __launch_bounds__(32) void attn_kernel(
    const _Float16* __restrict__ q16, const _Float16* __restrict__ k16,
    const _Float16* __restrict__ vT16, float* __restrict__ attn_out,
    float* __restrict__ attn_t_acc, _Float16* __restrict__ xa16)
{
    __shared__ __align__(16) float     sS[16][NTOK];
    __shared__ __align__(16) _Float16  sP[16][NTOK];

    const int lane = threadIdx.x & 31;
    int id = blockIdx.x;
    const int s = id % 36; id /= 36;          // 16-row strip index
    const int h = id % HEADS; id /= HEADS;
    const int b = id;
    const int row0 = s * 16;

    const size_t bh = (size_t)(b * HEADS + h);
    const _Float16* qbase = q16 + (bh * NTOK + row0) * HD;
    const _Float16* kbase = k16 + bh * NTOK * HD;
    const _Float16* vbase = vT16 + bh * HD * NTOK;

    const v16h aq0 = load_ab_frag(qbase + 0, HD);
    const v16h aq1 = load_ab_frag(qbase + 32, HD);

    const int cn = lane & 15;
    const int rb = (lane >> 4) << 3;
    const float scale = 0.125f;               // 1/sqrt(64)

    // ---- scores strip: 16 x 576
    for (int jt = 0; jt < NTOK / 16; ++jt) {
        v8f acc = (v8f){0,0,0,0,0,0,0,0};
        v16h b0 = load_ab_frag(kbase + (size_t)(jt * 16) * HD + 0, HD);
        v16h b1 = load_ab_frag(kbase + (size_t)(jt * 16) * HD + 32, HD);
        acc = wmma_f16(aq0, b0, acc);
        acc = wmma_f16(aq1, b1, acc);
#pragma unroll
        for (int e = 0; e < 8; ++e) sS[rb + e][jt * 16 + cn] = acc[e] * scale;
    }
    __syncthreads();

    // ---- softmax per row (lanes cooperate over 576 cols)
    for (int r = 0; r < 16; ++r) {
        float m = -3.4e38f;
        for (int c = lane; c < NTOK; c += 32) m = fmaxf(m, sS[r][c]);
        for (int o = 16; o; o >>= 1) m = fmaxf(m, __shfl_xor(m, o, 32));
        float sum = 0.f;
        for (int c = lane; c < NTOK; c += 32) {
            float e = __expf(sS[r][c] - m);
            sS[r][c] = e; sum += e;
        }
        for (int o = 16; o; o >>= 1) sum += __shfl_xor(sum, o, 32);
        const float inv = 1.0f / sum;
        for (int c = lane; c < NTOK; c += 32) {
            float p = sS[r][c] * inv;
            sS[r][c] = p;
            sP[r][c] = (_Float16)p;
        }
    }
    __syncthreads();

    // ---- materialize attention probabilities (dominant HBM traffic, ~510MB)
    float* aout = attn_out + (bh * NTOK + row0) * NTOK;
    for (int i = lane; i < 16 * NTOK; i += 32)
        aout[i] = sS[i / NTOK][i % NTOK];

    // ---- token scores: mean over template rows (0..63) and heads
    if (s < 4) {
        for (int c = LT + lane; c < NTOK; c += 32) {
            float v = 0.f;
#pragma unroll
            for (int r = 0; r < 16; ++r) v += sS[r][c];
            atomicAdd(&attn_t_acc[b * (2 * LS) + (c - LT)], v * (1.0f / (LT * HEADS)));
        }
    }

    // ---- O = P(16x576) x V(576x64), V stored transposed for contiguous B-frags
#pragma unroll 1
    for (int nt = 0; nt < HD / 16; ++nt) {
        v8f acc = (v8f){0,0,0,0,0,0,0,0};
        for (int kt = 0; kt < NTOK / 32; ++kt) {
            v16h a  = load_ab_frag(&sP[0][0] + kt * 32, NTOK);
            v16h bv = load_ab_frag(vbase + (size_t)(nt * 16) * NTOK + kt * 32, NTOK);
            acc = wmma_f16(a, bv, acc);
        }
#pragma unroll
        for (int e = 0; e < 8; ++e) {
            const int n = row0 + rb + e;
            const int c = h * HD + nt * 16 + cn;
            xa16[((size_t)(b * NTOK + n)) * DIMC + c] = (_Float16)acc[e];
        }
    }
}

// ---------------- top-k via 256-wide bitonic sort (descending, stable-ish) --------
__global__ __launch_bounds__(256) void topk_kernel(
    const float* __restrict__ attn_t, const int* __restrict__ gidx_ps,
    const int* __restrict__ gidx_s, float* __restrict__ keep_ps,
    float* __restrict__ keep_s, float* __restrict__ rem_ps,
    float* __restrict__ rem_s, int* __restrict__ perm)
{
    __shared__ float val[256];
    __shared__ int   idx[256];
    const int t = threadIdx.x;
    const int side = blockIdx.x & 1;
    const int b = blockIdx.x >> 1;

    val[t] = attn_t[b * (2 * LS) + side * LS + t];
    idx[t] = t;
    __syncthreads();

    for (int k = 2; k <= 256; k <<= 1) {
        for (int j = k >> 1; j > 0; j >>= 1) {
            const int ixj = t ^ j;
            if (ixj > t) {
                float va = val[t], vb = val[ixj];
                int   ia = idx[t], ib = idx[ixj];
                // descending by value, ascending by index on ties
                bool aFirst = (va > vb) || (va == vb && ia < ib);
                bool up = ((t & k) == 0);
                if (up ? !aFirst : aFirst) {
                    val[t] = vb; val[ixj] = va;
                    idx[t] = ib; idx[ixj] = ia;
                }
            }
            __syncthreads();
        }
    }

    const int p = idx[t];
    perm[(b * 2 + side) * LS + t] = p;
    const int* g = side ? gidx_s : gidx_ps;
    const float gi = (float)g[b * LS + p];
    if (side == 0) {
        if (t < KEEPN) keep_ps[b * KEEPN + t] = gi;
        else           rem_ps [b * REMN + (t - KEEPN)] = gi;
    } else {
        if (t < KEEPN) keep_s[b * KEEPN + t] = gi;
        else           rem_s [b * REMN + (t - KEEPN)] = gi;
    }
}

// ---------------- gather / concat reordered tokens --------------------------------
__global__ __launch_bounds__(256) void gather_kernel(
    const float* __restrict__ x1, const int* __restrict__ perm,
    float* __restrict__ x2)
{
    const int row = blockIdx.x;              // 0..B*NTOK-1
    const int b = row / NTOK, j = row % NTOK;
    int src;
    if (j < LT)            src = j;
    else if (j < LT + LS)  src = LT + perm[(b * 2 + 0) * LS + (j - LT)];
    else                   src = LT + LS + perm[(b * 2 + 1) * LS + (j - LT - LS)];
    const float* in  = x1 + ((size_t)b * NTOK + src) * DIMC;
    float*       out = x2 + (size_t)row * DIMC;
    out[threadIdx.x]       = in[threadIdx.x];
    out[threadIdx.x + 256] = in[threadIdx.x + 256];
    out[threadIdx.x + 512] = in[threadIdx.x + 512];
}

// ---------------- tiny helpers ----------------------------------------------------
__global__ void f32_to_f16_kernel(const float* __restrict__ in,
                                  _Float16* __restrict__ out, size_t n) {
    for (size_t i = (size_t)blockIdx.x * blockDim.x + threadIdx.x; i < n;
         i += (size_t)gridDim.x * blockDim.x)
        out[i] = (_Float16)in[i];
}
__global__ void i32_to_f32_kernel(const int* __restrict__ in,
                                  float* __restrict__ out, int n) {
    int i = blockIdx.x * blockDim.x + threadIdx.x;
    if (i < n) out[i] = (float)in[i];
}

// ---------------- host launch -----------------------------------------------------
extern "C" void kernel_launch(void* const* d_in, const int* in_sizes, int n_in,
                              void* d_out, int out_size, void* d_ws, size_t ws_size,
                              hipStream_t stream) {
    const float* x        = (const float*)d_in[0];
    const int*   gi_tmpl  = (const int*)d_in[1];
    const int*   gi_ps    = (const int*)d_in[2];
    const int*   gi_s     = (const int*)d_in[3];
    const float* ln1_w    = (const float*)d_in[4];
    const float* ln1_b    = (const float*)d_in[5];
    const float* qkv_w    = (const float*)d_in[6];
    const float* qkv_b    = (const float*)d_in[7];
    const float* proj_w   = (const float*)d_in[8];
    const float* proj_b   = (const float*)d_in[9];
    const float* ln2_w    = (const float*)d_in[10];
    const float* ln2_b    = (const float*)d_in[11];
    const float* fc1_w    = (const float*)d_in[12];
    const float* fc1_b    = (const float*)d_in[13];
    const float* fc2_w    = (const float*)d_in[14];
    const float* fc2_b    = (const float*)d_in[15];

    // ---- d_out layout (floats), reference return order
    float* out = (float*)d_out;
    const size_t o_x    = 0;
    const size_t o_tmpl = o_x    + (size_t)MROWS * DIMC;           // 14,155,776
    const size_t o_kps  = o_tmpl + (size_t)BATCH * LT;
    const size_t o_ks   = o_kps  + (size_t)BATCH * KEEPN;
    const size_t o_rps  = o_ks   + (size_t)BATCH * KEEPN;
    const size_t o_rs   = o_rps  + (size_t)BATCH * REMN;
    const size_t o_attn = o_rs   + (size_t)BATCH * REMN;

    // ---- workspace layout (bytes); g16 (fc1 out) aliases q/k/vT/xa
    char* ws = (char*)d_ws;
    const size_t szH16 = (size_t)MROWS * DIMC * 2;   // 28,311,552
    _Float16* q16   = (_Float16*)(ws + 0 * szH16);
    _Float16* k16   = (_Float16*)(ws + 1 * szH16);
    _Float16* vT16  = (_Float16*)(ws + 2 * szH16);
    _Float16* xa16  = (_Float16*)(ws + 3 * szH16);
    _Float16* g16   = (_Float16*)(ws + 0 * szH16);   // alias (MROWS x 3072 f16)
    _Float16* h16   = (_Float16*)(ws + 4 * szH16);   // LN out, reused for LN2
    float*    x1    = (float*)   (ws + 5 * szH16);   // f32 after proj residual
    float*    x2    = (float*)   (ws + 5 * szH16 + (size_t)MROWS * DIMC * 4);
    char*     wsw   = ws + 5 * szH16 + 2 * (size_t)MROWS * DIMC * 4;
    _Float16* wqkv16  = (_Float16*)(wsw);                      wsw += (size_t)3*DIMC*DIMC*2;
    _Float16* wproj16 = (_Float16*)(wsw);                      wsw += (size_t)DIMC*DIMC*2;
    _Float16* wfc1_16 = (_Float16*)(wsw);                      wsw += (size_t)MLPH*DIMC*2;
    _Float16* wfc2_16 = (_Float16*)(wsw);                      wsw += (size_t)DIMC*MLPH*2;
    float*    attn_t  = (float*)(wsw);                         wsw += (size_t)BATCH*2*LS*4;
    int*      perm    = (int*)(wsw);

    // 0) zero token-score accumulators
    hipMemsetAsync(attn_t, 0, (size_t)BATCH * 2 * LS * 4, stream);

    // 1) weights -> f16 (L2-resident thereafter)
    f32_to_f16_kernel<<<2048, 256, 0, stream>>>(qkv_w,  wqkv16,  (size_t)3*DIMC*DIMC);
    f32_to_f16_kernel<<<2048, 256, 0, stream>>>(proj_w, wproj16, (size_t)DIMC*DIMC);
    f32_to_f16_kernel<<<2048, 256, 0, stream>>>(fc1_w,  wfc1_16, (size_t)MLPH*DIMC);
    f32_to_f16_kernel<<<2048, 256, 0, stream>>>(fc2_w,  wfc2_16, (size_t)DIMC*MLPH);

    // 2) LN1
    ln_kernel<<<MROWS, 256, 0, stream>>>(x, ln1_w, ln1_b, h16);

    // 3) QKV GEMM: [18432 x 2304] = h16 x qkv_w^T, scatter to q/k/vT f16
    {
        EpiArgs ea{qkv_b, nullptr, nullptr, q16, k16, vT16};
        gemm_wmma_kernel<EPI_QKV><<<dim3(MROWS/256, (3*DIMC)/64), 256, 0, stream>>>(
            h16, wqkv16, MROWS, 3*DIMC, DIMC, ea);
    }

    // 4) fused attention (also writes attn tensor + token scores)
    attn_kernel<<<BATCH * HEADS * (NTOK/16), 32, 0, stream>>>(
        q16, k16, vT16, out + o_attn, attn_t, xa16);

    // 5) proj GEMM + residual: x1 = x + xa16 x proj_w^T
    {
        EpiArgs ea{proj_b, x, x1, nullptr, nullptr, nullptr};
        gemm_wmma_kernel<EPI_RESID32><<<dim3(MROWS/256, DIMC/64), 256, 0, stream>>>(
            xa16, wproj16, MROWS, DIMC, DIMC, ea);
    }

    // 6) top-k selection + index outputs
    topk_kernel<<<BATCH * 2, 256, 0, stream>>>(
        attn_t, gi_ps, gi_s, out + o_kps, out + o_ks, out + o_rps, out + o_rs, perm);

    // 7) reorder tokens -> x2 ; template indices -> out
    gather_kernel<<<MROWS, 256, 0, stream>>>(x1, perm, x2);
    i32_to_f32_kernel<<<(BATCH*LT + 255)/256, 256, 0, stream>>>(
        gi_tmpl, out + o_tmpl, BATCH*LT);

    // 8) LN2 (reuse h16)
    ln_kernel<<<MROWS, 256, 0, stream>>>(x2, ln2_w, ln2_b, h16);

    // 9) fc1 GEMM + exact GELU -> g16 (aliases freed q/k/vT/xa region)
    {
        EpiArgs ea{fc1_b, nullptr, nullptr, g16, nullptr, nullptr};
        gemm_wmma_kernel<EPI_GELU16><<<dim3(MROWS/256, MLPH/64), 256, 0, stream>>>(
            h16, wfc1_16, MROWS, MLPH, DIMC, ea);
    }

    // 10) fc2 GEMM + residual -> final x in d_out
    {
        EpiArgs ea{fc2_b, x2, out + o_x, nullptr, nullptr, nullptr};
        gemm_wmma_kernel<EPI_RESID32><<<dim3(MROWS/256, DIMC/64), 256, 0, stream>>>(
            g16, wfc2_16, MROWS, DIMC, MLPH, ea);
    }

    (void)n_in; (void)in_sizes; (void)out_size; (void)ws_size;
}
